// ScatterMoE_9414568313164
// MI455X (gfx1250) — compile-verified, matrix operand order checked by reading, not run on hardware
//
#include <hip/hip_runtime.h>
#include <hip/hip_bf16.h>
#include <math.h>

// ---------------------------------------------------------------------------
// ScatterMoE for MI455X (gfx1250): top-2 sparse MoE with bf16 WMMA GEMMs and
// async global->LDS double-buffered staging (ASYNCcnt pipeline).
//   T=4096 tokens, H=1024, I=4096, E=8, top-k=2
// d_out = [ y : T*H f32 | logits : T*E f32 ]
// Workspace (~244 MB): xb(bf16), w1t/w3t/w2t (bf16, K-contiguous), h(bf16),
//                      counts/tok/gate per-expert routing lists.
// ---------------------------------------------------------------------------

#define TDIM 4096
#define HDIM 1024
#define IDIM 4096
#define EDIM 8

#define LROW 40   // LDS row stride in bf16 (32 data + 8 pad => 80B, 16B-aligned)

typedef __attribute__((ext_vector_type(16))) __bf16 v16bf;
typedef __attribute__((ext_vector_type(8)))  float  v8f;

// ---- async global->LDS (CDNA5 GLOBAL_LOAD_ASYNC_TO_LDS_B128, ASYNCcnt) ----
#if __has_builtin(__builtin_amdgcn_global_load_async_to_lds_b128)
#define HAVE_ASYNC 1
#else
#define HAVE_ASYNC 0
#endif

typedef int v4i __attribute__((vector_size(16)));
typedef __attribute__((address_space(1))) v4i* as1_v4i;  // global int4*
typedef __attribute__((address_space(3))) v4i* as3_v4i;  // LDS int4*

__device__ __forceinline__ void cp16_async(__bf16* lds, const __bf16* g) {
#if HAVE_ASYNC
  __builtin_amdgcn_global_load_async_to_lds_b128(
      (as1_v4i)(g), (as3_v4i)(lds), 0, 0);
#else
  *(uint4*)lds = *(const uint4*)g;   // synchronous fallback, same barriers
#endif
}

#if HAVE_ASYNC
#if __has_builtin(__builtin_amdgcn_s_wait_asynccnt)
#define WAIT_ASYNC(n) __builtin_amdgcn_s_wait_asynccnt(n)
#else
#define WAIT_ASYNC(n) asm volatile("s_wait_asynccnt %0" ::"i"(n) : "memory")
#endif
#else
#define WAIT_ASYNC(n) do { } while (0)
#endif

// Fragment load (16 bf16/lane): lanes 0-15 hold K=[0..7],[16..23], lanes 16-31
// hold K=[8..15],[24..31] (ISA 7.12.2); caller bakes half*8 into the pointer.
__device__ __forceinline__ v16bf load_frag16(const __bf16* p) {
  union { uint4 q[2]; v16bf v; } u;
  u.q[0] = *(const uint4*)(p);
  u.q[1] = *(const uint4*)(p + 16);
  return u.v;
}

__device__ __forceinline__ v8f wmma_bf16(v16bf a, v16bf b, v8f c) {
  return __builtin_amdgcn_wmma_f32_16x16x32_bf16(
      false, a, false, b, (short)0, c, false, false);
}

// --------------------------- prep kernels ----------------------------------

__global__ void cast_x_kernel(const float* __restrict__ x,
                              __bf16* __restrict__ xb, int n4) {
  int i = blockIdx.x * blockDim.x + threadIdx.x;
  if (i >= n4) return;
  const float4 v = ((const float4*)x)[i];
  __bf16* o = xb + (size_t)i * 4;
  o[0] = (__bf16)v.x; o[1] = (__bf16)v.y; o[2] = (__bf16)v.z; o[3] = (__bf16)v.w;
}

// src: per-expert [R,C] f32 (blockIdx.z = expert) -> dst per-expert [C,R] bf16
__global__ void transpose_cast(const float* __restrict__ src,
                               __bf16* __restrict__ dst, int R, int C) {
  __shared__ float tile[32][33];
  src += (size_t)blockIdx.z * R * C;
  dst += (size_t)blockIdx.z * R * C;
  int c0 = blockIdx.x * 32, r0 = blockIdx.y * 32;
  int tx = threadIdx.x;
#pragma unroll
  for (int i = threadIdx.y; i < 32; i += 8)
    tile[i][tx] = src[(size_t)(r0 + i) * C + (c0 + tx)];
  __syncthreads();
#pragma unroll
  for (int i = threadIdx.y; i < 32; i += 8)
    dst[(size_t)(c0 + i) * R + (r0 + tx)] = (__bf16)tile[tx][i];
}

// --------------------------- router ----------------------------------------
__global__ void router_kernel(const float* __restrict__ x,
                              const float* __restrict__ rw,
                              float* __restrict__ logits_out,
                              int* __restrict__ counts,
                              int* __restrict__ tok,
                              float* __restrict__ gv) {
  int wave = (blockIdx.x * blockDim.x + threadIdx.x) >> 5;
  int lane = threadIdx.x & 31;
  if (wave >= TDIM) return;
  const int t = wave;

  float part[EDIM];
#pragma unroll
  for (int e = 0; e < EDIM; ++e) part[e] = 0.f;
  for (int k = lane; k < HDIM; k += 32) {
    float xv = x[(size_t)t * HDIM + k];
#pragma unroll
    for (int e = 0; e < EDIM; ++e) part[e] += xv * rw[e * HDIM + k];
  }
#pragma unroll
  for (int e = 0; e < EDIM; ++e)
#pragma unroll
    for (int off = 16; off; off >>= 1) part[e] += __shfl_xor(part[e], off);

  if (lane == 0) {
#pragma unroll
    for (int e = 0; e < EDIM; ++e) logits_out[(size_t)t * EDIM + e] = part[e];
    int a = 0;
#pragma unroll
    for (int e = 1; e < EDIM; ++e) if (part[e] > part[a]) a = e;
    int b = (a == 0) ? 1 : 0;
#pragma unroll
    for (int e = 0; e < EDIM; ++e)
      if (e != a && part[e] > part[b]) b = e;
    float d = __expf(part[b] - part[a]);       // stable: lb <= la
    float ga = 1.f / (1.f + d);
    float gb = d / (1.f + d);
    int pa = atomicAdd(&counts[a], 1);
    tok[a * TDIM + pa] = t; gv[a * TDIM + pa] = ga;
    int pb = atomicAdd(&counts[b], 1);
    tok[b * TDIM + pb] = t; gv[b * TDIM + pb] = gb;
  }
}

// --------------------------- expert GEMM pass 1 -----------------------------
// h[p,:] = silu(x[tok[p]] @ w1_e) * (x[tok[p]] @ w3_e) for compacted rows p.
// Block tile 128(M) x 64(N); async double-buffered LDS staging of A/B1/B3.
__global__ __launch_bounds__(256)
void moe_pass1(const __bf16* __restrict__ xb,
               const __bf16* __restrict__ w1t,   // [E][I][H] K-contiguous
               const __bf16* __restrict__ w3t,
               __bf16* __restrict__ h,           // [T][I]
               const int* __restrict__ counts,
               const int* __restrict__ tok, int e) {
  __shared__ __bf16 lA[2][128 * LROW];
  __shared__ __bf16 lB1[2][64 * LROW];
  __shared__ __bf16 lB3[2][64 * LROW];

  const int cnt = counts[e];
  const int Mbase = blockIdx.x * 128;
  if (Mbase >= cnt) return;                       // uniform early-exit
  const int Nbase = blockIdx.y * 64;

  const __bf16* w1e = w1t + (size_t)e * IDIM * HDIM;
  const __bf16* w3e = w3t + (size_t)e * IDIM * HDIM;
  const int* tk = tok + e * TDIM;
  const int tid = threadIdx.x;

  // A staging: thread -> row tid>>1 (0..127), two 16B chunks (token gather
  // resolved once, outside the K loop).
  const int arow = tid >> 1;
  const int akc = (tid & 1) * 2;
  const int pA = Mbase + arow;
  const int tA = (pA < cnt) ? tk[pA] : 0;
  const __bf16* agp = xb + (size_t)tA * HDIM + akc * 8;
  const int aofs = arow * LROW + akc * 8;
  // B staging: thread -> row tid>>2 (0..63), one 16B chunk per matrix.
  const int brow = tid >> 2, bkc = tid & 3;
  const __bf16* b1gp = w1e + (size_t)(Nbase + brow) * HDIM + bkc * 8;
  const __bf16* b3gp = w3e + (size_t)(Nbase + brow) * HDIM + bkc * 8;
  const int bofs = brow * LROW + bkc * 8;

#define STAGE1(bf, kk)                           \
  do {                                           \
    cp16_async(&lA[bf][aofs], agp + (kk));       \
    cp16_async(&lA[bf][aofs + 8], agp + (kk) + 8); \
    cp16_async(&lB1[bf][bofs], b1gp + (kk));     \
    cp16_async(&lB3[bf][bofs], b3gp + (kk));     \
  } while (0)

  const int lane = tid & 31, wave = tid >> 5;
  const int wm = wave & 3, wn = wave >> 2;
  const int half = lane >> 4, m16 = lane & 15;

  v8f acc1[2][2] = {}; v8f acc3[2][2] = {};
  STAGE1(0, 0);
  int buf = 0;
  for (int kk = 0; kk < HDIM; kk += 32) {
    if (kk + 32 < HDIM) {
      STAGE1(buf ^ 1, kk + 32);   // prefetch next K-step
      WAIT_ASYNC(4);              // leave only the 4 just-issued in flight
    } else {
      WAIT_ASYNC(0);
    }
    __syncthreads();

    v16bf a[2], b1f[2], b3f[2];
#pragma unroll
    for (int i = 0; i < 2; ++i)
      a[i] = load_frag16(&lA[buf][(wm * 32 + i * 16 + m16) * LROW + half * 8]);
#pragma unroll
    for (int j = 0; j < 2; ++j) {
      int n = (wn * 32 + j * 16 + m16) * LROW + half * 8;
      b1f[j] = load_frag16(&lB1[buf][n]);
      b3f[j] = load_frag16(&lB3[buf][n]);
    }
#pragma unroll
    for (int i = 0; i < 2; ++i)
#pragma unroll
      for (int j = 0; j < 2; ++j) {
        acc1[i][j] = wmma_bf16(a[i], b1f[j], acc1[i][j]);
        acc3[i][j] = wmma_bf16(a[i], b3f[j], acc3[i][j]);
      }
    __syncthreads();              // safe to overwrite buf next iteration
    buf ^= 1;
  }
#undef STAGE1

  // C layout: VGPR r, lanes 0-15 -> M=r, lanes 16-31 -> M=8+r; N = lane&15.
#pragma unroll
  for (int i = 0; i < 2; ++i)
#pragma unroll
    for (int j = 0; j < 2; ++j) {
      int row0 = Mbase + wm * 32 + i * 16 + half * 8;
      int col  = Nbase + wn * 32 + j * 16 + m16;
#pragma unroll
      for (int r = 0; r < 8; ++r) {
        float s1 = acc1[i][j][r], s3 = acc3[i][j][r];
        float sig = __builtin_amdgcn_rcpf(1.f + __expf(-s1));  // v_rcp_f32
        h[(size_t)(row0 + r) * IDIM + col] = (__bf16)(s1 * sig * s3);
      }
    }
}

// --------------------------- expert GEMM pass 2 -----------------------------
// y[tok[p]] += gate[p] * (h[p] @ w2_e).  Block tile 128(M) x 128(N); each of
// 8 waves owns 32x64 (2x4 WMMA accumulators). Async LDS staging of A and B.
__global__ __launch_bounds__(256)
void moe_pass2(const __bf16* __restrict__ h,     // [T][I]
               const __bf16* __restrict__ w2t,   // [E][H][I] K-contiguous
               float* __restrict__ y,            // [T][H]
               const int* __restrict__ counts,
               const int* __restrict__ tok,
               const float* __restrict__ gv, int e) {
  __shared__ __bf16 lA[2][128 * LROW];
  __shared__ __bf16 lB[2][128 * LROW];

  const int cnt = counts[e];
  const int Mbase = blockIdx.x * 128;
  if (Mbase >= cnt) return;
  const int Nbase = blockIdx.y * 128;

  const __bf16* w2e = w2t + (size_t)e * HDIM * IDIM;
  const int* tk = tok + e * TDIM;
  const float* gve = gv + e * TDIM;
  const int tid = threadIdx.x;

  // Staging: thread -> row tid>>1 (0..127), two 16B chunks for A and for B.
  const int srow = tid >> 1;
  const int skc = (tid & 1) * 2;
  const __bf16* agp = h + (size_t)(Mbase + srow) * IDIM + skc * 8;  // rows compacted
  const __bf16* bgp = w2e + (size_t)(Nbase + srow) * IDIM + skc * 8;
  const int sofs = srow * LROW + skc * 8;

#define STAGE2(bf, kk)                           \
  do {                                           \
    cp16_async(&lA[bf][sofs], agp + (kk));       \
    cp16_async(&lA[bf][sofs + 8], agp + (kk) + 8); \
    cp16_async(&lB[bf][sofs], bgp + (kk));       \
    cp16_async(&lB[bf][sofs + 8], bgp + (kk) + 8); \
  } while (0)

  const int lane = tid & 31, wave = tid >> 5;
  const int wm = wave & 3, wn = wave >> 2;        // 4 x 2 waves
  const int half = lane >> 4, m16 = lane & 15;

  v8f acc[2][4] = {};
  STAGE2(0, 0);
  int buf = 0;
  for (int kk = 0; kk < IDIM; kk += 32) {
    if (kk + 32 < IDIM) {
      STAGE2(buf ^ 1, kk + 32);
      WAIT_ASYNC(4);
    } else {
      WAIT_ASYNC(0);
    }
    __syncthreads();

    v16bf a[2], b[4];
#pragma unroll
    for (int i = 0; i < 2; ++i)
      a[i] = load_frag16(&lA[buf][(wm * 32 + i * 16 + m16) * LROW + half * 8]);
#pragma unroll
    for (int j = 0; j < 4; ++j)
      b[j] = load_frag16(&lB[buf][(wn * 64 + j * 16 + m16) * LROW + half * 8]);
#pragma unroll
    for (int i = 0; i < 2; ++i)
#pragma unroll
      for (int j = 0; j < 4; ++j)
        acc[i][j] = wmma_bf16(a[i], b[j], acc[i][j]);
    __syncthreads();
    buf ^= 1;
  }
#undef STAGE2

#pragma unroll
  for (int i = 0; i < 2; ++i)
#pragma unroll
    for (int j = 0; j < 4; ++j) {
      int row0 = Mbase + wm * 32 + i * 16 + half * 8;
      int col  = Nbase + wn * 64 + j * 16 + m16;
#pragma unroll
      for (int r = 0; r < 8; ++r) {
        int p = row0 + r;
        if (p < cnt) {
          int   t = tk[p];
          float g = gve[p];
          y[(size_t)t * HDIM + col] += g * acc[i][j][r];
        }
      }
    }
}

// --------------------------- launch ----------------------------------------

extern "C" void kernel_launch(void* const* d_in, const int* in_sizes, int n_in,
                              void* d_out, int out_size, void* d_ws, size_t ws_size,
                              hipStream_t stream) {
  (void)in_sizes; (void)n_in; (void)out_size; (void)ws_size;
  const float* x  = (const float*)d_in[0];   // [B,S,H]
  const float* rw = (const float*)d_in[1];   // [E,H]
  const float* w1 = (const float*)d_in[2];   // [E,H,I]
  const float* w2 = (const float*)d_in[3];   // [E,I,H]
  const float* w3 = (const float*)d_in[4];   // [E,H,I]
  float* out = (float*)d_out;                // y [T,H] ++ logits [T,E]

  char* ws = (char*)d_ws;
  size_t off = 0;
  auto take = [&](size_t bytes) -> char* {
    char* p = ws + off;
    off = (off + bytes + 255) & ~(size_t)255;
    return p;
  };
  __bf16* xb   = (__bf16*)take((size_t)TDIM * HDIM * 2);
  __bf16* w1t  = (__bf16*)take((size_t)EDIM * HDIM * IDIM * 2);
  __bf16* w3t  = (__bf16*)take((size_t)EDIM * HDIM * IDIM * 2);
  __bf16* w2t  = (__bf16*)take((size_t)EDIM * HDIM * IDIM * 2);
  __bf16* hbuf = (__bf16*)take((size_t)TDIM * IDIM * 2);
  int*    counts = (int*)take(EDIM * sizeof(int));
  int*    tok    = (int*)take((size_t)EDIM * TDIM * sizeof(int));
  float*  gv     = (float*)take((size_t)EDIM * TDIM * sizeof(float));

  (void)hipMemsetAsync(out, 0, (size_t)TDIM * HDIM * sizeof(float), stream);
  (void)hipMemsetAsync(counts, 0, EDIM * sizeof(int), stream);

  cast_x_kernel<<<(TDIM * HDIM / 4 + 255) / 256, 256, 0, stream>>>(
      x, xb, TDIM * HDIM / 4);
  dim3 tb(32, 8);
  transpose_cast<<<dim3(IDIM / 32, HDIM / 32, EDIM), tb, 0, stream>>>(w1, w1t, HDIM, IDIM);
  transpose_cast<<<dim3(IDIM / 32, HDIM / 32, EDIM), tb, 0, stream>>>(w3, w3t, HDIM, IDIM);
  transpose_cast<<<dim3(HDIM / 32, IDIM / 32, EDIM), tb, 0, stream>>>(w2, w2t, IDIM, HDIM);

  router_kernel<<<TDIM / 8, 256, 0, stream>>>(
      x, rw, out + (size_t)TDIM * HDIM, counts, tok, gv);

  for (int e = 0; e < EDIM; ++e) {
    moe_pass1<<<dim3(TDIM / 128, IDIM / 64), 256, 0, stream>>>(
        xb, w1t, w3t, hbuf, counts, tok, e);
    moe_pass2<<<dim3(TDIM / 128, HDIM / 128), 256, 0, stream>>>(
        hbuf, w2t, out, counts, tok, gv, e);
  }
}
